// RecommendationLSTM_13864154432028
// MI455X (gfx1250) — matrix operand order repaired and checked by reading, not compiled, not run on hardware
//
#include <hip/hip_runtime.h>
#include <hip/hip_bf16.h>

// ---------------------------------------------------------------------------
// RecommendationLSTM fused for gfx1250 (MI455X)
//   Kernel 1: gather review/product embeddings -> X_bf16 [T, B, 160] (K pad)
//   Kernel 2: fused LSTM: per step  gates = [X_t | h] @ [W_ih | W_hh]^T + b
//             via v_wmma_f32_16x16x32_bf16, weights resident in LDS (bf16),
//             h/c resident per-WG for all 512 steps, final FC + sigmoid.
//   Scheduling: per K-chunk, the next chunk's 8 ds_load_b128 are pinned ahead
//   of the current chunk's 4 WMMAs via sched_group_barrier so DS latency is
//   hidden behind XDL execution (waits become dscnt<=8, not 0).
// ---------------------------------------------------------------------------

typedef __attribute__((ext_vector_type(16))) __bf16 v16bf;
typedef __attribute__((ext_vector_type(8)))  float  v8f;

#define BATCH 256
#define TSTEPS 512
#define EMB 100
#define PEMB 50
#define INP 150
#define KIN 160          // padded input K (multiple of 32)
#define HID 128
#define KTOT (KIN + HID) // 288 = 9 chunks of 32
#define G4H 512          // 4*HID
#define WSTRIDE 296      // LDS row stride (bf16) for weights: 16B aligned, conflict-free
#define HSTRIDE 136      // LDS row stride (bf16) for h

#define SGB_DS_READ 0x100
#define SGB_WMMA    0x008

__device__ __forceinline__ unsigned short f32_to_bf16(float f) {
  unsigned int u = __float_as_uint(f);
  u += 0x7FFFu + ((u >> 16) & 1u);      // round-to-nearest-even
  return (unsigned short)(u >> 16);
}
__device__ __forceinline__ float bf16_to_f32(unsigned short h) {
  return __uint_as_float(((unsigned int)h) << 16);
}
__device__ __forceinline__ float sigmoidf_fast(float x) {
  return 1.0f / (1.0f + __expf(-x));
}
// tanh(x) = 1 - 2/(exp(2x)+1): one v_exp_f32 + rcp, robust at both infinities.
__device__ __forceinline__ float tanhf_fast(float x) {
  return 1.0f - 2.0f / (__expf(2.0f * x) + 1.0f);
}

// Workgroup barrier that only orders LDS (DScnt), leaving global loads in
// flight: s_wait_dscnt 0 ; s_barrier_signal -1 ; s_barrier_wait -1.
__device__ __forceinline__ void wg_barrier_lds() {
  asm volatile("s_wait_dscnt 0x0\n\t"
               "s_barrier_signal -1\n\t"
               "s_barrier_wait -1" ::: "memory");
}

union BF16x16 { unsigned int u[8]; v16bf v; };

// Load a 16-element bf16 fragment as two 16-byte chunks (p0: 8 bf16, p1: 8 bf16).
template <typename PT>
__device__ __forceinline__ v16bf load_frag(PT p0, PT p1) {
  BF16x16 r;
  uint4 a = *(const uint4*)p0;
  uint4 b = *(const uint4*)p1;
  r.u[0]=a.x; r.u[1]=a.y; r.u[2]=a.z; r.u[3]=a.w;
  r.u[4]=b.x; r.u[5]=b.y; r.u[6]=b.z; r.u[7]=b.w;
  return r.v;
}

// ---------------------------------------------------------------------------
// Kernel 1: build X_bf16[t*B + b][0..159] = [emb[review[b][t]] | pemb[product[b]] | 0pad]
// One thread per bf16 pair (80 pairs per row).
// ---------------------------------------------------------------------------
__global__ void embed_pack_kernel(const int* __restrict__ review,
                                  const int* __restrict__ product,
                                  const float* __restrict__ emb,
                                  const float* __restrict__ pemb,
                                  unsigned short* __restrict__ X) {
  int gid = blockIdx.x * blockDim.x + threadIdx.x;
  if (gid >= BATCH * TSTEPS * (KIN / 2)) return;
  int p   = gid % (KIN / 2);
  int row = gid / (KIN / 2);          // row = t*BATCH + b
  int b   = row & (BATCH - 1);
  int t   = row >> 8;
  int k0  = p * 2;
  float v0 = 0.0f, v1 = 0.0f;
  if (k0 < EMB) {
    int idx = review[b * TSTEPS + t];
    const float* e = emb + (size_t)idx * EMB;
    v0 = e[k0]; v1 = e[k0 + 1];
  } else if (k0 < INP) {
    int pi = product[b];
    const float* e = pemb + (size_t)pi * PEMB;
    v0 = e[k0 - EMB]; v1 = e[k0 - EMB + 1];
  }
  unsigned int pk = (unsigned int)f32_to_bf16(v0) |
                    ((unsigned int)f32_to_bf16(v1) << 16);
  *(unsigned int*)(X + (size_t)row * KIN + k0) = pk;
}

// ---------------------------------------------------------------------------
// Kernel 2: fused LSTM. 16 blocks (one per 16 batch rows), 256 threads = 8 waves.
// Wave w owns hidden jtile w (16 hidden units): gate tiles i,f,g,o.
// LDS: Wl[512][296] bf16 = [W_ih pad | W_hh], Hl[16][136] bf16 = h state.
// ---------------------------------------------------------------------------
__global__ void __launch_bounds__(256, 1)
lstm_fused_kernel(const unsigned short* __restrict__ X,
                  const float* __restrict__ W_ih,
                  const float* __restrict__ W_hh,
                  const float* __restrict__ b_ih,
                  const float* __restrict__ b_hh,
                  const float* __restrict__ fc_w,
                  const float* __restrict__ fc_b,
                  float* __restrict__ out) {
  extern __shared__ unsigned short smem[];
  unsigned short* Wl = smem;                       // G4H x WSTRIDE
  unsigned short* Hl = smem + G4H * WSTRIDE;       // 16 x HSTRIDE

  const int tid = threadIdx.x;

  // ---- stage weights into LDS as bf16: cols [0,160)=W_ih (pad 150..159=0), [160,288)=W_hh
  for (int idx = tid; idx < G4H * (KTOT / 2); idx += 256) {
    int g  = idx / (KTOT / 2);
    int k0 = (idx - g * (KTOT / 2)) * 2;
    float v0, v1;
    if (k0 < KIN) {
      v0 = (k0     < INP) ? W_ih[g * INP + k0]     : 0.0f;
      v1 = (k0 + 1 < INP) ? W_ih[g * INP + k0 + 1] : 0.0f;
    } else {
      int kk = k0 - KIN;
      v0 = W_hh[g * HID + kk];
      v1 = W_hh[g * HID + kk + 1];
    }
    unsigned int pk = (unsigned int)f32_to_bf16(v0) |
                      ((unsigned int)f32_to_bf16(v1) << 16);
    *(unsigned int*)(Wl + g * WSTRIDE + k0) = pk;
  }
  // ---- zero h state
  for (int idx = tid; idx < 16 * HSTRIDE / 2; idx += 256)
    *(unsigned int*)(Hl + idx * 2) = 0u;

  const int lane = tid & 31;
  const int wv   = tid >> 5;                 // jtile 0..7
  const int nlo  = lane & 15;                // M/N index within tile
  const int kbA  = (lane < 16) ? 0 : 16;     // A-fragment K base (contiguous K, matches B)
  const int kbB  = (lane < 16) ? 0 : 16;     // B-fragment K base (contiguous K, matches A)
  const int jcol = wv * 16 + nlo;            // hidden unit column handled by this lane

  float bias0 = b_ih[0 * HID + jcol] + b_hh[0 * HID + jcol];
  float bias1 = b_ih[1 * HID + jcol] + b_hh[1 * HID + jcol];
  float bias2 = b_ih[2 * HID + jcol] + b_hh[2 * HID + jcol];
  float bias3 = b_ih[3 * HID + jcol] + b_hh[3 * HID + jcol];

  // Weight rows (LDS) for this lane's four gate tiles.
  const unsigned short* wr0 = Wl + (0 * HID + jcol) * WSTRIDE + kbB;
  const unsigned short* wr1 = Wl + (1 * HID + jcol) * WSTRIDE + kbB;
  const unsigned short* wr2 = Wl + (2 * HID + jcol) * WSTRIDE + kbB;
  const unsigned short* wr3 = Wl + (3 * HID + jcol) * WSTRIDE + kbB;

  v8f c = {};                                // cell-state tile (f32, resident)
  const int brow = blockIdx.x * 16 + nlo;    // batch row this lane feeds into A
  const unsigned short* xbase = X + (size_t)brow * KIN + kbA;

  __syncthreads();

  for (int t = 0; t < TSTEPS; ++t) {
    // ---- A fragments: 5 chunks from X (global), 4 chunks from h (LDS)
    const unsigned short* xr = xbase + (size_t)t * (BATCH * KIN);
    v16bf A[9];
    #pragma unroll
    for (int kc = 0; kc < 5; ++kc) {
      const unsigned short* p = xr + kc * 32;
      A[kc] = load_frag(p, p + 8);
    }
    #pragma unroll
    for (int kc = 0; kc < 4; ++kc) {
      const unsigned short* p = Hl + nlo * HSTRIDE + kc * 32 + kbA;
      A[5 + kc] = load_frag(p, p + 8);
    }
    if (t + 1 < TSTEPS)
      __builtin_prefetch(xr + BATCH * KIN, 0, 3);   // -> global_prefetch_b8
    wg_barrier_lds();  // h(t-1) consumed by all waves; global X loads stay in flight

    // ---- 4 gate tiles, K pipelined: double-buffered B frags, 4 accums interleaved
    v8f a0 = { bias0, bias0, bias0, bias0, bias0, bias0, bias0, bias0 };
    v8f a1 = { bias1, bias1, bias1, bias1, bias1, bias1, bias1, bias1 };
    v8f a2 = { bias2, bias2, bias2, bias2, bias2, bias2, bias2, bias2 };
    v8f a3 = { bias3, bias3, bias3, bias3, bias3, bias3, bias3, bias3 };

    // Prologue: B frags for kc=0 (8 ds_load_b128).
    v16bf B0 = load_frag(wr0, wr0 + 8);
    v16bf B1 = load_frag(wr1, wr1 + 8);
    v16bf B2 = load_frag(wr2, wr2 + 8);
    v16bf B3 = load_frag(wr3, wr3 + 8);
    __builtin_amdgcn_sched_group_barrier(SGB_DS_READ, 8, 0);

    #pragma unroll
    for (int kc = 0; kc < 9; ++kc) {
      v16bf N0, N1, N2, N3;
      if (kc < 8) {                       // issue next K-chunk's 8 ds_load_b128
        int off = (kc + 1) * 32;
        N0 = load_frag(wr0 + off, wr0 + off + 8);
        N1 = load_frag(wr1 + off, wr1 + off + 8);
        N2 = load_frag(wr2 + off, wr2 + off + 8);
        N3 = load_frag(wr3 + off, wr3 + off + 8);
      }
      a0 = __builtin_amdgcn_wmma_f32_16x16x32_bf16(false, A[kc], false, B0, (short)0, a0, false, false);
      a1 = __builtin_amdgcn_wmma_f32_16x16x32_bf16(false, A[kc], false, B1, (short)0, a1, false, false);
      a2 = __builtin_amdgcn_wmma_f32_16x16x32_bf16(false, A[kc], false, B2, (short)0, a2, false, false);
      a3 = __builtin_amdgcn_wmma_f32_16x16x32_bf16(false, A[kc], false, B3, (short)0, a3, false, false);
      if (kc < 8) {
        B0 = N0; B1 = N1; B2 = N2; B3 = N3;
        // Pin schedule: [8 DS reads (kc+1)] ahead of [4 WMMAs (kc)].
        __builtin_amdgcn_sched_group_barrier(SGB_DS_READ, 8, 0);
      }
      __builtin_amdgcn_sched_group_barrier(SGB_WMMA, 4, 0);
    }

    // ---- elementwise LSTM cell (torch gate order i,f,g,o) + publish h(t) as bf16
    #pragma unroll
    for (int r = 0; r < 8; ++r) {
      float iv = sigmoidf_fast(a0[r]);
      float fv = sigmoidf_fast(a1[r]);
      float gv = tanhf_fast(a2[r]);
      float ov = sigmoidf_fast(a3[r]);
      float cn = fv * c[r] + iv * gv;
      c[r] = cn;
      float hv = ov * tanhf_fast(cn);
      int row = (lane < 16) ? r : (r + 8);
      Hl[row * HSTRIDE + jcol] = f32_to_bf16(hv);
    }
    wg_barrier_lds();   // h(t) visible before next step's A loads
  }

  // ---- final FC + sigmoid: lanes 0..15 of the block each handle one batch row
  if (tid < 16) {
    float accum = fc_b[0];
    #pragma unroll 8
    for (int j = 0; j < HID; ++j)
      accum += bf16_to_f32(Hl[tid * HSTRIDE + j]) * fc_w[j];
    out[blockIdx.x * 16 + tid] = 1.0f / (1.0f + __expf(-accum));
  }
}

// ---------------------------------------------------------------------------
extern "C" void kernel_launch(void* const* d_in, const int* in_sizes, int n_in,
                              void* d_out, int out_size, void* d_ws, size_t ws_size,
                              hipStream_t stream) {
  const int*   review  = (const int*)  d_in[0];
  const int*   product = (const int*)  d_in[1];
  const float* emb     = (const float*)d_in[2];
  const float* pemb    = (const float*)d_in[3];
  const float* W_ih    = (const float*)d_in[4];
  const float* W_hh    = (const float*)d_in[5];
  const float* b_ih    = (const float*)d_in[6];
  const float* b_hh    = (const float*)d_in[7];
  const float* fc_w    = (const float*)d_in[8];
  const float* fc_b    = (const float*)d_in[9];
  float* out = (float*)d_out;

  unsigned short* X = (unsigned short*)d_ws;   // T*B*160 bf16 = 40 MB

  {
    int total  = BATCH * TSTEPS * (KIN / 2);
    int blocks = (total + 255) / 256;
    embed_pack_kernel<<<blocks, 256, 0, stream>>>(review, product, emb, pemb, X);
  }
  {
    size_t shmem = (size_t)(G4H * WSTRIDE + 16 * HSTRIDE) * sizeof(unsigned short);
    lstm_fused_kernel<<<BATCH / 16, 256, shmem, stream>>>(
        X, W_ih, W_hh, b_ih, b_hh, fc_w, fc_b, out);
  }
}
